// DendriticMLP_22282290332395
// MI455X (gfx1250) — matrix operand order repaired and checked by reading, not compiled, not run on hardware
//
#include <hip/hip_runtime.h>
#include <hip/hip_bf16.h>

typedef __attribute__((ext_vector_type(16))) __bf16 v16bf;
typedef __attribute__((ext_vector_type(8)))  __bf16 v8bf;
typedef __attribute__((ext_vector_type(8)))  float  v8f;
typedef __attribute__((ext_vector_type(4)))  int    v4i;

#define BSZ    4096
#define D_IN   1024
#define HIDDEN 2048
#define N_SEG  10
#define D_CTX  1024
#define D_OUT  1024
#define KWIN   102

#if __has_builtin(__builtin_amdgcn_global_load_async_to_lds_b128) && \
    __has_builtin(__builtin_amdgcn_s_wait_asynccnt)
#define HAVE_ASYNC_LDS 1
#else
#define HAVE_ASYNC_LDS 0
#endif

typedef __attribute__((address_space(1))) v4i as1_v4i;  // prints as "__device__"
typedef __attribute__((address_space(3))) v4i as3_v4i;  // LDS

// ---------------------------------------------------------------------------
// fp32 -> bf16 conversion (grid-stride)
// ---------------------------------------------------------------------------
__global__ void cvt_f32_bf16(const float* __restrict__ src,
                             __bf16* __restrict__ dst, int n) {
  int i = blockIdx.x * blockDim.x + threadIdx.x;
  int stride = gridDim.x * blockDim.x;
  for (; i < n; i += stride) dst[i] = (__bf16)src[i];
}

// ---------------------------------------------------------------------------
// WMMA fragment builders (bf16 16x16x32, wave32).
// A 16x32: lane L = row (L&15); halves 0-7 = k [kb,kb+8), 8-15 = [kb+16,kb+24),
//          kb = (L>>4)*8  -> two 16B loads at p and p+16.
// B 32x16: lane L = col (L&15); same k split as A, kb = (L>>4)*8
//          -> two 16B loads at p and p+16.  Works for global or LDS pointers.
// ---------------------------------------------------------------------------
__device__ __forceinline__ v16bf frag_a(const __bf16* p) {
  v8bf lo = *(const v8bf*)(p);
  v8bf hi = *(const v8bf*)(p + 16);
  v16bf r;
#pragma unroll
  for (int e = 0; e < 8; ++e) { r[e] = lo[e]; r[e + 8] = hi[e]; }
  return r;
}
__device__ __forceinline__ v16bf frag_b(const __bf16* p) {
  v8bf lo = *(const v8bf*)(p);
  v8bf hi = *(const v8bf*)(p + 16);
  v16bf r;
#pragma unroll
  for (int e = 0; e < 8; ++e) { r[e] = lo[e]; r[e + 8] = hi[e]; }
  return r;
}

#define WMMA_BF16(A, B, C)                                             \
  __builtin_amdgcn_wmma_f32_16x16x32_bf16(false, (A), false, (B),      \
                                          (short)0, (C), false, false)

// 16-byte chunk: global -> LDS (async DMA when available)
__device__ __forceinline__ void lds_fill16(__bf16* ldst, const __bf16* gsrc) {
#if HAVE_ASYNC_LDS
  __builtin_amdgcn_global_load_async_to_lds_b128(
      (as1_v4i*)gsrc, (as3_v4i*)ldst, 0, 0);
#else
  *(v8bf*)ldst = *(const v8bf*)gsrc;
#endif
}
__device__ __forceinline__ void lds_fill_wait() {
#if HAVE_ASYNC_LDS
  __builtin_amdgcn_s_wait_asynccnt(0);
#endif
}

// ---------------------------------------------------------------------------
// Fused dendritic layer.
// Block = 8 waves sharing one 16-wide hidden-column tile (n0), covering 128
// batch rows (16 per wave). Shared B operands (10 segment tiles, then the W
// tile) are staged in LDS, double-buffered, cooperatively filled.
//   Phase 1: dend[s] = CTX @ SEG[:,s,:]^T (10 accs) -> sel = abs-max select
//   Phase 2: y = X @ W^T
//   Epilogue: act = (y + b) * sigmoid(sel)
// ---------------------------------------------------------------------------
#define TILE_H  (16 * 32)          // halves per 16x32 B tile (1 KB)

__global__ __launch_bounds__(256)
void dendrite_fused(const __bf16* __restrict__ X,
                    const __bf16* __restrict__ CTX,
                    const __bf16* __restrict__ W,
                    const float*  __restrict__ bias,
                    const __bf16* __restrict__ SEG,
                    float* __restrict__ act, int kdim) {
  __shared__ __bf16 smem[2][N_SEG * TILE_H];   // 2 x 10 KB

  const int tid  = threadIdx.x;
  const int lane = tid & 31;
  const int wave = tid >> 5;
  const int n0 = blockIdx.x * 16;              // hidden-col tile (shared)
  const int m0 = blockIdx.y * 128 + wave * 16; // batch-row tile (per wave)
  const int rsel = lane & 15;
  const int ksel = lane >> 4;

  const int arow = m0 + rsel;   // A-fragment row (batch sample)

  // cooperative fill of the 10 segment B tiles for K-chunk k0
  auto fill_seg = [&](int buf, int k0) {
#pragma unroll 1
    for (int c = tid; c < N_SEG * 16 * 4; c += 256) {
      const int t = c >> 6;          // segment
      const int r = (c >> 2) & 15;   // hidden row within tile
      const int q = c & 3;           // 16B chunk within row
      const __bf16* g =
          SEG + ((size_t)(n0 + r) * N_SEG + t) * D_CTX + k0 + q * 8;
      lds_fill16(&smem[buf][(t * 16 + r) * 32 + q * 8], g);
    }
  };
  // cooperative fill of the W B-tile for K-chunk k0 (tile 0 of buffer)
  auto fill_w = [&](int buf, int k0) {
    for (int c = tid; c < 16 * 4; c += 256) {
      const int r = (c >> 2) & 15;
      const int q = c & 3;
      const __bf16* g = W + (size_t)(n0 + r) * kdim + k0 + q * 8;
      lds_fill16(&smem[buf][r * 32 + q * 8], g);
    }
  };

  // ---- Phase 1: dendrite segments --------------------------------------
  v8f acc_s[N_SEG];
#pragma unroll
  for (int s = 0; s < N_SEG; ++s) acc_s[s] = (v8f){};

  fill_seg(0, 0);
  {
    const __bf16* pctx = CTX + (size_t)arow * D_CTX + ksel * 8;
    int buf = 0;
    for (int k0 = 0; k0 < D_CTX; k0 += 32) {
      lds_fill_wait();
      __syncthreads();                       // buffer `buf` ready
      if (k0 + 32 < D_CTX) fill_seg(buf ^ 1, k0 + 32);
      v16bf ac = frag_a(pctx);
      const __bf16* bb = &smem[buf][rsel * 32 + ksel * 8];
      // two-deep pipeline on the LDS B fragments
      v16bf bcur = frag_b(bb);
#pragma unroll
      for (int s = 0; s < N_SEG; ++s) {
        v16bf bnext;
        if (s + 1 < N_SEG) bnext = frag_b(bb + (s + 1) * TILE_H);
        acc_s[s] = WMMA_BF16(ac, bcur, acc_s[s]);
        bcur = bnext;
      }
      pctx += 32;
      buf ^= 1;
    }
  }

  // collapse 10 accumulators -> abs-max-selected value per element
  v8f sel;
#pragma unroll
  for (int i = 0; i < 8; ++i) {
    float sv = acc_s[0][i];
    float best = fabsf(sv);
#pragma unroll
    for (int s = 1; s < N_SEG; ++s) {
      float v = acc_s[s][i];
      float a = fabsf(v);
      if (a > best) { best = a; sv = v; }
    }
    sel[i] = sv;
  }

  // ---- Phase 2: feed-forward GEMM --------------------------------------
  v8f acc0 = {}, acc1 = {};
  {
    __syncthreads();                         // phase-1 readers done
    fill_w(0, 0);
    const __bf16* px = X + (size_t)arow * kdim + ksel * 8;
    int buf = 0;
    for (int k0 = 0; k0 < kdim; k0 += 32) {
      lds_fill_wait();
      __syncthreads();
      if (k0 + 32 < kdim) fill_w(buf ^ 1, k0 + 32);
      v16bf a0 = frag_a(px);
      v16bf b0 = frag_b(&smem[buf][rsel * 32 + ksel * 8]);
      if (k0 & 32)
        acc1 = WMMA_BF16(a0, b0, acc1);
      else
        acc0 = WMMA_BF16(a0, b0, acc0);
      px += 32;
      buf ^= 1;
    }
  }

  // ---- Epilogue: bias + sigmoid gating ---------------------------------
  // C/D layout: VGPR i of lane L -> (row = m0 + i + 8*(L>>4), col = n0+(L&15))
  const int ocol = n0 + rsel;
  const int orow0 = m0 + ksel * 8;
  const float bv = bias[ocol];
#pragma unroll
  for (int i = 0; i < 8; ++i) {
    float gate = 1.0f / (1.0f + __expf(-sel[i]));
    float y = acc0[i] + acc1[i] + bv;
    act[(size_t)(orow0 + i) * HIDDEN + ocol] = y * gate;
  }
}

// ---------------------------------------------------------------------------
// KWinners: keep top-K per row of HIDDEN, zero the rest; emit bf16.
// Exact k-th largest via binary search on the monotonic uint key of floats.
// ---------------------------------------------------------------------------
__device__ __forceinline__ unsigned fkey(float f) {
  unsigned u = __float_as_uint(f);
  return (u & 0x80000000u) ? ~u : (u | 0x80000000u);
}

__global__ __launch_bounds__(256)
void kwinners_kernel(const float* __restrict__ act,
                     __bf16* __restrict__ hout) {
  __shared__ float row[HIDDEN];
  __shared__ int cnt;
  const int b = blockIdx.x;
  const float* r = act + (size_t)b * HIDDEN;
  for (int j = threadIdx.x; j < HIDDEN; j += blockDim.x) row[j] = r[j];
  __syncthreads();

  unsigned lo = 0u, hi = 0xFFFFFFFFu;
  for (int it = 0; it < 32 && lo < hi; ++it) {
    unsigned mid = lo + (((hi - lo) >> 1) + 1u);
    if (threadIdx.x == 0) cnt = 0;
    __syncthreads();
    int c = 0;
    for (int j = threadIdx.x; j < HIDDEN; j += blockDim.x)
      c += (fkey(row[j]) >= mid) ? 1 : 0;
    atomicAdd(&cnt, c);
    __syncthreads();
    if (cnt >= KWIN) lo = mid; else hi = mid - 1u;
    __syncthreads();
  }
  for (int j = threadIdx.x; j < HIDDEN; j += blockDim.x) {
    float v = row[j];
    hout[(size_t)b * HIDDEN + j] =
        (fkey(v) >= lo) ? (__bf16)v : (__bf16)0.0f;
  }
}

// ---------------------------------------------------------------------------
// Output GEMM: out = H @ Wout^T + b_out  (bf16 WMMA, fp32 out, 2 chains)
// ---------------------------------------------------------------------------
__global__ __launch_bounds__(256)
void out_gemm(const __bf16* __restrict__ H,
              const __bf16* __restrict__ WO,
              const float* __restrict__ bias,
              float* __restrict__ out) {
  const int lane = threadIdx.x & 31;
  const int wave = threadIdx.x >> 5;
  const int n0 = blockIdx.x * 128 + wave * 16;
  const int m0 = blockIdx.y * 16;
  const int rsel = lane & 15;
  const int ksel = lane >> 4;

  v8f acc0 = {}, acc1 = {};
  const __bf16* ph = H + (size_t)(m0 + rsel) * HIDDEN + ksel * 8;
  const __bf16* pw = WO + (size_t)(n0 + rsel) * HIDDEN + ksel * 8;
  for (int k0 = 0; k0 < HIDDEN; k0 += 64) {
    v16bf a0 = frag_a(ph);
    v16bf b0 = frag_b(pw);
    v16bf a1 = frag_a(ph + 32);
    v16bf b1 = frag_b(pw + 32);
    __builtin_prefetch(ph + 64, 0, 0);
    __builtin_prefetch(pw + 64, 0, 0);
    acc0 = WMMA_BF16(a0, b0, acc0);
    acc1 = WMMA_BF16(a1, b1, acc1);
    ph += 64;
    pw += 64;
  }
  const int ocol = n0 + rsel;
  const int orow0 = m0 + ksel * 8;
  const float bv = bias[ocol];
#pragma unroll
  for (int i = 0; i < 8; ++i)
    out[(size_t)(orow0 + i) * D_OUT + ocol] = acc0[i] + acc1[i] + bv;
}

// ---------------------------------------------------------------------------
// Launch
// ---------------------------------------------------------------------------
extern "C" void kernel_launch(void* const* d_in, const int* in_sizes, int n_in,
                              void* d_out, int out_size, void* d_ws,
                              size_t ws_size, hipStream_t stream) {
  const float* x    = (const float*)d_in[0];
  const float* ctx  = (const float*)d_in[1];
  const float* w1   = (const float*)d_in[2];
  const float* b1   = (const float*)d_in[3];
  const float* seg1 = (const float*)d_in[4];
  const float* w2   = (const float*)d_in[5];
  const float* b2   = (const float*)d_in[6];
  const float* seg2 = (const float*)d_in[7];
  const float* wout = (const float*)d_in[8];
  const float* bout = (const float*)d_in[9];
  float* out = (float*)d_out;

  char* p = (char*)d_ws;
  auto alloc = [&](size_t bytes) -> void* {
    void* r = (void*)p;
    p += (bytes + 255) & ~(size_t)255;
    return r;
  };
  __bf16* xb    = (__bf16*)alloc((size_t)BSZ * D_IN * 2);
  __bf16* ctxb  = (__bf16*)alloc((size_t)BSZ * D_CTX * 2);
  __bf16* w1b   = (__bf16*)alloc((size_t)HIDDEN * D_IN * 2);
  __bf16* seg1b = (__bf16*)alloc((size_t)HIDDEN * N_SEG * D_CTX * 2);
  __bf16* w2b   = (__bf16*)alloc((size_t)HIDDEN * HIDDEN * 2);
  __bf16* seg2b = (__bf16*)alloc((size_t)HIDDEN * N_SEG * D_CTX * 2);
  __bf16* woutb = (__bf16*)alloc((size_t)D_OUT * HIDDEN * 2);
  float*  act   = (float*)alloc((size_t)BSZ * HIDDEN * 4);
  __bf16* h1b   = (__bf16*)alloc((size_t)BSZ * HIDDEN * 2);
  __bf16* h2b   = (__bf16*)alloc((size_t)BSZ * HIDDEN * 2);

  const int CT = 256;
  auto cvt = [&](const float* s, __bf16* d, size_t n) {
    int blocks = (int)((n + CT * 8 - 1) / (CT * 8));
    cvt_f32_bf16<<<blocks, CT, 0, stream>>>(s, d, (int)n);
  };
  cvt(x,    xb,    (size_t)BSZ * D_IN);
  cvt(ctx,  ctxb,  (size_t)BSZ * D_CTX);
  cvt(w1,   w1b,   (size_t)HIDDEN * D_IN);
  cvt(seg1, seg1b, (size_t)HIDDEN * N_SEG * D_CTX);
  cvt(w2,   w2b,   (size_t)HIDDEN * HIDDEN);
  cvt(seg2, seg2b, (size_t)HIDDEN * N_SEG * D_CTX);
  cvt(wout, woutb, (size_t)D_OUT * HIDDEN);

  dim3 gridH(HIDDEN / 16, BSZ / 128);
  dendrite_fused<<<gridH, 256, 0, stream>>>(xb, ctxb, w1b, b1, seg1b, act, D_IN);
  kwinners_kernel<<<BSZ, 256, 0, stream>>>(act, h1b);
  dendrite_fused<<<gridH, 256, 0, stream>>>(h1b, ctxb, w2b, b2, seg2b, act, HIDDEN);
  kwinners_kernel<<<BSZ, 256, 0, stream>>>(act, h2b);
  dim3 gridO(D_OUT / 128, BSZ / 16);
  out_gemm<<<gridO, 256, 0, stream>>>(h2b, woutb, bout, out);
}
